// SuperGraphConstruction_42279658062320
// MI455X (gfx1250) — matrix operand order repaired and checked by reading, not compiled, not run on hardware
//
#include <hip/hip_runtime.h>

typedef __attribute__((ext_vector_type(16))) _Float16 v16h;
typedef __attribute__((ext_vector_type(8)))  _Float16 v8h;
typedef __attribute__((ext_vector_type(8)))  float    v8f;

constexpr int N_NODES = 100000;
constexpr int C_CLUST = 20000;
constexpr int LAT = 128;
constexpr int HID = 512;
constexpr int EMB = 12;
constexpr int ES  = 160000;
constexpr int EB  = 800000;

#define DEV __device__ __forceinline__

// ---------------- WMMA helpers (ISA 7.12.2 layouts, wave32) ----------------

DEV v8f wmma_f16(v16h a, v16h b, v8f c) {
  // emits v_wmma_f32_16x16x32_f16
  return __builtin_amdgcn_wmma_f32_16x16x32_f16(
      /*neg_a=*/false, a, /*neg_b=*/false, b,
      /*c_mod=*/(short)0, c, /*reuse_a=*/false, /*reuse_b=*/false);
}

// A (16x32 f16) from row-major f32 global. Per-lane halves are two contiguous
// 8-element runs: k = kb..kb+7 and k = kb+16..kb+23  ->  4x global_load_b128.
DEV v16h frag_a_g(const float* __restrict__ X, int ld, int row0, int k0, int maxrow) {
  const int lane = threadIdx.x & 31;
  int r = row0 + (lane & 15);
  if (r > maxrow) r = maxrow;               // clamp tail rows (stores are guarded)
  const float* p = X + (size_t)r * ld + k0 + ((lane >> 4) << 3);
  const float4 x0 = ((const float4*)p)[0];
  const float4 x1 = ((const float4*)p)[1];
  const float4 x2 = ((const float4*)(p + 16))[0];
  const float4 x3 = ((const float4*)(p + 16))[1];
  v16h a;
  a[0]  = (_Float16)x0.x; a[1]  = (_Float16)x0.y; a[2]  = (_Float16)x0.z; a[3]  = (_Float16)x0.w;
  a[4]  = (_Float16)x1.x; a[5]  = (_Float16)x1.y; a[6]  = (_Float16)x1.z; a[7]  = (_Float16)x1.w;
  a[8]  = (_Float16)x2.x; a[9]  = (_Float16)x2.y; a[10] = (_Float16)x2.z; a[11] = (_Float16)x2.w;
  a[12] = (_Float16)x3.x; a[13] = (_Float16)x3.y; a[14] = (_Float16)x3.z; a[15] = (_Float16)x3.w;
  return a;
}

// A (16x32 f16) from f16 LDS: 2x ds_load_b128 per lane.
DEV v16h frag_a_lds(const _Float16* __restrict__ S, int ld, int row0, int k0) {
  const int lane = threadIdx.x & 31;
  const _Float16* p = S + (size_t)(row0 + (lane & 15)) * ld + k0 + ((lane >> 4) << 3);
  const v8h lo = *(const v8h*)p;
  const v8h hi = *(const v8h*)(p + 16);
  v16h a;
#pragma unroll
  for (int i = 0; i < 8; ++i) { a[i] = lo[i]; a[8 + i] = hi[i]; }
  return a;
}

// B fragment from pre-packed fragment-major weights:
// P[((nt*ktiles + kt)*32 + lane)*16 + idx]  ->  one contiguous 32B load per lane.
DEV v16h frag_bp(const _Float16* __restrict__ P, int ktiles, int nt, int kt) {
  const int lane = threadIdx.x & 31;
  return ((const v16h*)P)[(size_t)(nt * ktiles + kt) * 32 + lane];
}

DEV v8f bias_acc(const float* __restrict__ B, int n0) {
  const float x = B[n0 + (threadIdx.x & 15)];   // C/D col = lane%16
  v8f a;
#pragma unroll
  for (int i = 0; i < 8; ++i) a[i] = x;
  return a;
}

// D (16x16 f32): lane holds col n=L%16; VGPR v holds row m = v + 8*(L/16)
template <int ACT>   // 0 = tanh, 1 = relu
DEV void store_d_lds(_Float16* __restrict__ S, int ld, int row0, int n0, v8f d) {
  const int lane = threadIdx.x & 31;
  const int n = n0 + (lane & 15);
#pragma unroll
  for (int v = 0; v < 8; ++v) {
    const int m = row0 + v + ((lane >> 4) << 3);
    float x = d[v];
    x = (ACT == 0) ? tanhf(x) : fmaxf(x, 0.0f);
    S[(size_t)m * ld + n] = (_Float16)x;
  }
}

// ---------------- weight repack: W[K][N] f32 -> fragment-major f16 ----------------
// out index i = ((nt*(K/32) + kt)*32 + lane)*16 + (2v+p)
// maps to    k = kt*32 + (lane>>4)*8 + ((2v+p)&7) + ((2v+p)>>3)*16 ,  n = nt*16 + (lane&15)
__global__ void k_pack_b(const float* __restrict__ W, _Float16* __restrict__ P, int K, int N) {
  const int i = blockIdx.x * blockDim.x + threadIdx.x;
  if (i >= K * N) return;
  const int idx    = i & 15;
  const int lane   = (i >> 4) & 31;
  const int tile   = i >> 9;
  const int ktiles = K >> 5;
  const int kt = tile % ktiles;
  const int nt = tile / ktiles;
  const int k = (kt << 5) + ((lane >> 4) << 3) + (idx & 7) + ((idx >> 3) << 4);
  const int n = (nt << 4) + (lane & 15);
  P[i] = (_Float16)W[(size_t)k * N + n];
}

// ---------------- fused MLP kernels (64 threads = 2 waves, 32 rows / block) ----------------

__global__ __launch_bounds__(64) void k_mlp_emb(
    const float* __restrict__ X,
    const _Float16* __restrict__ W1, const float* __restrict__ B1,
    const _Float16* __restrict__ W2, const float* __restrict__ B2,
    const float* __restrict__ W3, const float* __restrict__ B3,
    const int* __restrict__ clusters,
    float* __restrict__ emb_out,
    float* __restrict__ csum, float* __restrict__ ccnt)
{
  __shared__ __align__(16) _Float16 H1[32 * HID];   // 32 KB
  __shared__ __align__(16) _Float16 H2[32 * HID];   // 32 KB
  const int wave  = threadIdx.x >> 5;
  const int lrow0 = wave * 16;
  const int grow0 = blockIdx.x * 32 + lrow0;

  // layer 1: tanh(X @ W1 + B1), K = LAT. Two n-tiles per iter share A fragments.
  for (int t = 0; t < HID / 16; t += 2) {
    v8f acc0 = bias_acc(B1, t * 16);
    v8f acc1 = bias_acc(B1, t * 16 + 16);
#pragma unroll
    for (int kt = 0; kt < LAT / 32; ++kt) {
      v16h a  = frag_a_g(X, LAT, grow0, kt * 32, N_NODES - 1);
      v16h b0 = frag_bp(W1, LAT / 32, t,     kt);
      v16h b1 = frag_bp(W1, LAT / 32, t + 1, kt);
      acc0 = wmma_f16(a, b0, acc0);
      acc1 = wmma_f16(a, b1, acc1);
    }
    store_d_lds<0>(H1, HID, lrow0, t * 16,      acc0);
    store_d_lds<0>(H1, HID, lrow0, t * 16 + 16, acc1);
  }
  __syncthreads();

  // layer 2: tanh(H1 @ W2 + B2), K = HID
  for (int t = 0; t < HID / 16; t += 2) {
    v8f acc0 = bias_acc(B2, t * 16);
    v8f acc1 = bias_acc(B2, t * 16 + 16);
    for (int kt = 0; kt < HID / 32; ++kt) {
      if (kt + 1 < HID / 32)
        __builtin_prefetch((const _Float16*)W2 + (size_t)(t * (HID / 32) + kt + 1) * 512, 0, 1);
      v16h a  = frag_a_lds(H1, HID, lrow0, kt * 32);
      v16h b0 = frag_bp(W2, HID / 32, t,     kt);
      v16h b1 = frag_bp(W2, HID / 32, t + 1, kt);
      acc0 = wmma_f16(a, b0, acc0);
      acc1 = wmma_f16(a, b1, acc1);
    }
    store_d_lds<0>(H2, HID, lrow0, t * 16,      acc0);
    store_d_lds<0>(H2, HID, lrow0, t * 16 + 16, acc1);
  }
  __syncthreads();

  // layer 3 (HID -> EMB=12): thread pairs split the 12 outputs; all 64 lanes active
  const int r  = threadIdx.x >> 1;          // 0..31 local row
  const int j0 = (threadIdx.x & 1) * 6;     // 0 or 6
  const int g  = blockIdx.x * 32 + r;
  float e[6];
#pragma unroll
  for (int j = 0; j < 6; ++j) e[j] = B3[j0 + j];
  for (int k = 0; k < HID; ++k) {
    const float h = (float)H2[r * HID + k];
    const float* w = W3 + (size_t)k * EMB + j0;
#pragma unroll
    for (int j = 0; j < 6; ++j) e[j] = fmaf(h, w[j], e[j]);
  }
  float ss = 0.f;
#pragma unroll
  for (int j = 0; j < 6; ++j) ss += e[j] * e[j];
  ss += __shfl_xor(ss, 1);                  // pair lanes combine the L2 norm
  const float inv = 1.0f / fmaxf(sqrtf(ss), 1e-12f);
  if (g < N_NODES) {
    const int c = clusters[g];
#pragma unroll
    for (int j = 0; j < 6; ++j) {
      const float v = e[j] * inv;
      emb_out[(size_t)g * EMB + j0 + j] = v;
      atomicAdd(&csum[(size_t)c * EMB + j0 + j], v);
    }
    if (j0 == 0) atomicAdd(&ccnt[c], 1.0f);
  }
}

__global__ __launch_bounds__(64) void k_mlp_msg(
    const float* __restrict__ X,
    const _Float16* __restrict__ W1, const float* __restrict__ B1,
    const _Float16* __restrict__ W2, const float* __restrict__ B2,
    float* __restrict__ nm)
{
  __shared__ __align__(16) _Float16 H1[32 * HID];   // 32 KB
  const int wave  = threadIdx.x >> 5;
  const int lrow0 = wave * 16;
  const int grow0 = blockIdx.x * 32 + lrow0;
  const int lane  = threadIdx.x & 31;

  for (int t = 0; t < HID / 16; t += 2) {
    v8f acc0 = bias_acc(B1, t * 16);
    v8f acc1 = bias_acc(B1, t * 16 + 16);
#pragma unroll
    for (int kt = 0; kt < LAT / 32; ++kt) {
      v16h a  = frag_a_g(X, LAT, grow0, kt * 32, N_NODES - 1);
      v16h b0 = frag_bp(W1, LAT / 32, t,     kt);
      v16h b1 = frag_bp(W1, LAT / 32, t + 1, kt);
      acc0 = wmma_f16(a, b0, acc0);
      acc1 = wmma_f16(a, b1, acc1);
    }
    store_d_lds<1>(H1, HID, lrow0, t * 16,      acc0);
    store_d_lds<1>(H1, HID, lrow0, t * 16 + 16, acc1);
  }
  __syncthreads();

  for (int t = 0; t < LAT / 16; t += 2) {
    v8f acc0 = bias_acc(B2, t * 16);
    v8f acc1 = bias_acc(B2, t * 16 + 16);
    for (int kt = 0; kt < HID / 32; ++kt) {
      v16h a  = frag_a_lds(H1, HID, lrow0, kt * 32);
      v16h b0 = frag_bp(W2, HID / 32, t,     kt);
      v16h b1 = frag_bp(W2, HID / 32, t + 1, kt);
      acc0 = wmma_f16(a, b0, acc0);
      acc1 = wmma_f16(a, b1, acc1);
    }
#pragma unroll
    for (int v = 0; v < 8; ++v) {
      const int g = grow0 + v + ((lane >> 4) << 3);
      if (g < N_NODES) {
        nm[(size_t)g * LAT + t * 16 +      (lane & 15)] = fmaxf(acc0[v], 0.0f);
        nm[(size_t)g * LAT + t * 16 + 16 + (lane & 15)] = fmaxf(acc1[v], 0.0f);
      }
    }
  }
}

__global__ __launch_bounds__(64) void k_superedge(
    const float* __restrict__ SN, const int* __restrict__ SG,
    const _Float16* __restrict__ W1, const float* __restrict__ B1,
    const _Float16* __restrict__ W2, const float* __restrict__ B2,
    float* __restrict__ out)
{
  __shared__ __align__(16) _Float16 Xt[32 * 2 * LAT];  // gathered concat input, 16 KB
  __shared__ __align__(16) _Float16 H [32 * HID];      // 32 KB
  const int e0 = blockIdx.x * 32;
  const int lane = threadIdx.x & 31;

  // stage se_in = concat(SN[sg0], SN[sg1]) as f16
  for (int idx = threadIdx.x; idx < 32 * 2 * LAT; idx += 64) {
    const int r = idx >> 8;       // / 256
    const int c = idx & 255;
    int e = e0 + r; if (e >= ES) e = ES - 1;
    const int node = (c < LAT) ? SG[e] : SG[ES + e];
    Xt[idx] = (_Float16)SN[(size_t)node * LAT + (c & (LAT - 1))];
  }
  __syncthreads();

  const int wave  = threadIdx.x >> 5;
  const int lrow0 = wave * 16;

  for (int t = 0; t < HID / 16; t += 2) {
    v8f acc0 = bias_acc(B1, t * 16);
    v8f acc1 = bias_acc(B1, t * 16 + 16);
#pragma unroll
    for (int kt = 0; kt < (2 * LAT) / 32; ++kt) {
      v16h a  = frag_a_lds(Xt, 2 * LAT, lrow0, kt * 32);
      v16h b0 = frag_bp(W1, (2 * LAT) / 32, t,     kt);
      v16h b1 = frag_bp(W1, (2 * LAT) / 32, t + 1, kt);
      acc0 = wmma_f16(a, b0, acc0);
      acc1 = wmma_f16(a, b1, acc1);
    }
    store_d_lds<1>(H, HID, lrow0, t * 16,      acc0);
    store_d_lds<1>(H, HID, lrow0, t * 16 + 16, acc1);
  }
  __syncthreads();

  for (int t = 0; t < LAT / 16; t += 2) {
    v8f acc0 = bias_acc(B2, t * 16);
    v8f acc1 = bias_acc(B2, t * 16 + 16);
    for (int kt = 0; kt < HID / 32; ++kt) {
      v16h a  = frag_a_lds(H, HID, lrow0, kt * 32);
      v16h b0 = frag_bp(W2, HID / 32, t,     kt);
      v16h b1 = frag_bp(W2, HID / 32, t + 1, kt);
      acc0 = wmma_f16(a, b0, acc0);
      acc1 = wmma_f16(a, b1, acc1);
    }
#pragma unroll
    for (int v = 0; v < 8; ++v) {
      const int m = lrow0 + v + ((lane >> 4) << 3);
      const int e = e0 + m;
      if (e < ES) {
        out[(size_t)e * LAT + t * 16 +      (lane & 15)] = fmaxf(acc0[v], 0.0f);
        out[(size_t)e * LAT + t * 16 + 16 + (lane & 15)] = fmaxf(acc1[v], 0.0f);
      }
    }
  }
}

// ---------------- elementwise / segment kernels ----------------

__global__ void k_means(const float* __restrict__ csum, const float* __restrict__ ccnt,
                        float* __restrict__ means) {
  const int c = blockIdx.x * blockDim.x + threadIdx.x;
  if (c >= C_CLUST) return;
  const float inv = 1.0f / fmaxf(ccnt[c], 1.0f);
  float v[EMB]; float ss = 0.f;
#pragma unroll
  for (int j = 0; j < EMB; ++j) { v[j] = csum[(size_t)c * EMB + j] * inv; ss += v[j] * v[j]; }
  const float s = 1.0f / fmaxf(sqrtf(ss), 1e-12f);
#pragma unroll
  for (int j = 0; j < EMB; ++j) means[(size_t)c * EMB + j] = v[j] * s;
}

// lik[e] = dot(A[IA[e]], B[IB[e]]) over EMB dims; accumulates sum/sumsq for batchnorm
__global__ void k_dot12(const float* __restrict__ A, const int* __restrict__ IA,
                        const float* __restrict__ B, const int* __restrict__ IB,
                        int n, float* __restrict__ lik, float* __restrict__ stats)
{
  const int e = blockIdx.x * blockDim.x + threadIdx.x;
  float x = 0.f;
  if (e < n) {
    const float* a = A + (size_t)IA[e] * EMB;
    const float* b = B + (size_t)IB[e] * EMB;
#pragma unroll
    for (int j = 0; j < EMB; ++j) x = fmaf(a[j], b[j], x);
    lik[e] = x;
  }
  float s = x, s2 = x * x;
#pragma unroll
  for (int o = 16; o > 0; o >>= 1) {
    s  += __shfl_down(s,  o);
    s2 += __shfl_down(s2, o);
  }
  if ((threadIdx.x & 31) == 0) { atomicAdd(&stats[0], s); atomicAdd(&stats[1], s2); }
}

__global__ void k_sigmoid_bn(const float* __restrict__ lik, const float* __restrict__ stats,
                             const float* __restrict__ g, const float* __restrict__ b,
                             int n, float* __restrict__ out)
{
  const int e = blockIdx.x * blockDim.x + threadIdx.x;
  if (e >= n) return;
  const float m   = stats[0] / (float)n;
  const float var = fmaxf(stats[1] / (float)n - m * m, 0.f);
  const float xh  = (lik[e] - m) * rsqrtf(var + 1e-5f) * g[0] + b[0];
  out[e] = 1.0f / (1.0f + expf(-xh));
}

__global__ void k_exp_bn_scatter(const float* __restrict__ lik, const float* __restrict__ stats,
                                 const float* __restrict__ g, const float* __restrict__ b,
                                 const int* __restrict__ src,
                                 float* __restrict__ bgw, float* __restrict__ denom)
{
  const int e = blockIdx.x * blockDim.x + threadIdx.x;
  if (e >= EB) return;
  const float m   = stats[0] / (float)EB;
  const float var = fmaxf(stats[1] / (float)EB - m * m, 0.f);
  const float w   = expf((lik[e] - m) * rsqrtf(var + 1e-5f) * g[0] + b[0]);
  bgw[e] = w;
  atomicAdd(&denom[src[e]], w);
}

__global__ void k_bg_norm(const float* __restrict__ bgw, const float* __restrict__ denom,
                          const int* __restrict__ src, float* __restrict__ out)
{
  const int e = blockIdx.x * blockDim.x + threadIdx.x;
  if (e >= EB) return;
  out[e] = bgw[e] / (1e-12f + denom[src[e]]);
}

__global__ void k_scatter_sn(const float* __restrict__ nm, const int* __restrict__ bsrc,
                             const int* __restrict__ bdst, const float* __restrict__ bw,
                             float* __restrict__ SN)
{
  const size_t i = (size_t)blockIdx.x * blockDim.x + threadIdx.x;
  if (i >= (size_t)EB * LAT) return;
  const int e = (int)(i >> 7);      // / LAT
  const int d = (int)(i & (LAT - 1));
  atomicAdd(&SN[(size_t)bdst[e] * LAT + d],
            nm[(size_t)bsrc[e] * LAT + d] * bw[e]);
}

// ---------------- launch ----------------

extern "C" void kernel_launch(void* const* d_in, const int* in_sizes, int n_in,
                              void* d_out, int out_size, void* d_ws, size_t ws_size,
                              hipStream_t stream)
{
  (void)in_sizes; (void)n_in; (void)out_size; (void)ws_size;

  const float* embedded = (const float*)d_in[0];
  const float* encoded  = (const float*)d_in[1];
  const float* Wc1 = (const float*)d_in[2];  const float* bc1 = (const float*)d_in[3];
  const float* Wc2 = (const float*)d_in[4];  const float* bc2 = (const float*)d_in[5];
  const float* Wc3 = (const float*)d_in[6];  const float* bc3 = (const float*)d_in[7];
  const float* Wn1 = (const float*)d_in[8];  const float* cn1 = (const float*)d_in[9];
  const float* Wn2 = (const float*)d_in[10]; const float* cn2 = (const float*)d_in[11];
  const float* We1 = (const float*)d_in[12]; const float* ce1 = (const float*)d_in[13];
  const float* We2 = (const float*)d_in[14]; const float* ce2 = (const float*)d_in[15];
  const float* sg_gamma = (const float*)d_in[16];
  const float* sg_beta  = (const float*)d_in[17];
  const float* bg_gamma = (const float*)d_in[18];
  const float* bg_beta  = (const float*)d_in[19];
  const int* clusters   = (const int*)d_in[20];
  const int* sg         = (const int*)d_in[21];   // [2, ES]
  const int* bg_src     = (const int*)d_in[22];
  const int* bg_dst     = (const int*)d_in[23];

  float* out     = (float*)d_out;
  float* emb_out = out;                                   // [N, EMB]
  float* sn_out  = emb_out + (size_t)N_NODES * EMB;       // [C, LAT]
  float* se_out  = sn_out  + (size_t)C_CLUST * LAT;       // [ES, LAT]
  float* bw_out  = se_out  + (size_t)ES * LAT;            // [EB]
  float* sew_out = bw_out  + (size_t)EB;                  // [ES]

  char* ws = (char*)d_ws;
  size_t off = 0;
  auto take = [&](size_t bytes) -> void* {
    void* p = ws + off;
    off = (off + bytes + 255) & ~(size_t)255;
    return p;
  };
  _Float16* W1c = (_Float16*)take((size_t)LAT * HID * 2);
  _Float16* W2c = (_Float16*)take((size_t)HID * HID * 2);
  _Float16* W1n = (_Float16*)take((size_t)LAT * HID * 2);
  _Float16* W2n = (_Float16*)take((size_t)HID * LAT * 2);
  _Float16* W1e = (_Float16*)take((size_t)2 * LAT * HID * 2);
  _Float16* W2e = (_Float16*)take((size_t)HID * LAT * 2);
  float* csum  = (float*)take((size_t)C_CLUST * EMB * 4);
  float* ccnt  = (float*)take((size_t)C_CLUST * 4);
  float* means = (float*)take((size_t)C_CLUST * EMB * 4);
  float* stats = (float*)take(4 * 4);                     // [sg_sum, sg_sq, bg_sum, bg_sq]
  float* sglik = (float*)take((size_t)ES * 4);
  float* bglik = (float*)take((size_t)EB * 4);
  float* bgw   = (float*)take((size_t)EB * 4);
  float* denom = (float*)take((size_t)N_NODES * 4);
  float* nmsg  = (float*)take((size_t)N_NODES * LAT * 4);

  // zero accumulators (graph-capture-safe)
  hipMemsetAsync(csum,  0, (size_t)C_CLUST * EMB * 4, stream);
  hipMemsetAsync(ccnt,  0, (size_t)C_CLUST * 4,       stream);
  hipMemsetAsync(stats, 0, 16,                        stream);
  hipMemsetAsync(denom, 0, (size_t)N_NODES * 4,       stream);
  hipMemsetAsync(sn_out,0, (size_t)C_CLUST * LAT * 4, stream);

  // f32 -> f16 fragment-major weight repack
  k_pack_b<<<(LAT * HID + 255) / 256, 256, 0, stream>>>(Wc1, W1c, LAT, HID);
  k_pack_b<<<(HID * HID + 255) / 256, 256, 0, stream>>>(Wc2, W2c, HID, HID);
  k_pack_b<<<(LAT * HID + 255) / 256, 256, 0, stream>>>(Wn1, W1n, LAT, HID);
  k_pack_b<<<(HID * LAT + 255) / 256, 256, 0, stream>>>(Wn2, W2n, HID, LAT);
  k_pack_b<<<(2 * LAT * HID + 255) / 256, 256, 0, stream>>>(We1, W1e, 2 * LAT, HID);
  k_pack_b<<<(HID * LAT + 255) / 256, 256, 0, stream>>>(We2, W2e, HID, LAT);

  const int mlpBlocks = (N_NODES + 31) / 32;

  // embeddings MLP (WMMA) + cluster segment-sum
  k_mlp_emb<<<mlpBlocks, 64, 0, stream>>>(embedded, W1c, bc1, W2c, bc2, Wc3, bc3,
                                          clusters, emb_out, csum, ccnt);
  k_means<<<(C_CLUST + 255) / 256, 256, 0, stream>>>(csum, ccnt, means);

  // super-graph edge weights
  k_dot12<<<(ES + 255) / 256, 256, 0, stream>>>(means, sg, means, sg + ES, ES, sglik, stats + 0);
  k_sigmoid_bn<<<(ES + 255) / 256, 256, 0, stream>>>(sglik, stats + 0, sg_gamma, sg_beta, ES, sew_out);

  // bipartite edge weights (softmax over edges per source node)
  k_dot12<<<(EB + 255) / 256, 256, 0, stream>>>(emb_out, bg_src, means, bg_dst, EB, bglik, stats + 2);
  k_exp_bn_scatter<<<(EB + 255) / 256, 256, 0, stream>>>(bglik, stats + 2, bg_gamma, bg_beta,
                                                         bg_src, bgw, denom);
  k_bg_norm<<<(EB + 255) / 256, 256, 0, stream>>>(bgw, denom, bg_src, bw_out);

  // node messages MLP (WMMA), weighted scatter into supernodes
  k_mlp_msg<<<mlpBlocks, 64, 0, stream>>>(encoded, W1n, cn1, W2n, cn2, nmsg);
  k_scatter_sn<<<(int)(((size_t)EB * LAT + 255) / 256), 256, 0, stream>>>(nmsg, bg_src, bg_dst,
                                                                          bw_out, sn_out);

  // super-edge MLP (WMMA)
  k_superedge<<<(ES + 31) / 32, 64, 0, stream>>>(sn_out, sg, W1e, ce1, W2e, ce2, se_out);
}
